// MyTopoAgent_27857157882206
// MI455X (gfx1250) — compile-verified, matrix operand order checked by reading, not run on hardware
//
#include <hip/hip_runtime.h>
#include <hip/hip_bf16.h>

#define BB     4
#define NNODE  10000
#define MM     (BB * NNODE)      // 40000 nodes
#define EE     640000            // raw edges
#define ETOT   (EE + MM)         // + self loops = 680000
#define CH     7

typedef __attribute__((ext_vector_type(16))) __bf16 v16bf;
typedef __attribute__((ext_vector_type(8)))  float  v8f;

// ---------- helpers ----------------------------------------------------------

__device__ __forceinline__ void edge_st(const int* __restrict__ ei, int e, int& s, int& t) {
  if (e < EE) { s = ei[e]; t = ei[EE + e]; }
  else        { s = e - EE; t = e - EE; }          // appended self loops
}

// order-preserving float->uint key for atomicMax-based segment max
__device__ __forceinline__ unsigned fkey(float f) {
  unsigned u = __float_as_uint(f);
  return (u & 0x80000000u) ? ~u : (u | 0x80000000u);
}
__device__ __forceinline__ float funkey(unsigned k) {
  unsigned u = (k & 0x80000000u) ? (k & 0x7FFFFFFFu) : ~k;
  return __uint_as_float(u);
}

__device__ __forceinline__ float lrelu(float v, float s) { return v > 0.f ? v : s * v; }

// ---------- bf16 WMMA GEMM: out[M,128] = A[M,128] @ W[128,128] + bias --------
// block = 128 threads = 4 waves; each wave owns a 16-row tile, all 8 col tiles.
__global__ __launch_bounds__(128)
void k_gemm128(const float* __restrict__ A, const float* __restrict__ W,
               const float* __restrict__ bias, float* __restrict__ out) {
  __shared__ __bf16 ldsw[128 * 130];               // W^T, padded to kill bank conflicts
  const int tid = threadIdx.x;
  for (int k = 0; k < 128; ++k)                    // coalesced read of W row k
    ldsw[tid * 130 + k] = (__bf16)W[k * 128 + tid];
  __syncthreads();

  const int wave = tid >> 5;
  const int lane = tid & 31;
  const int m16  = lane & 15;                      // row within A tile / col within D tile
  const int hi   = lane >> 4;                      // half selector (K offset for A/B, +8 rows for D)
  const int rowbase = blockIdx.x * 64 + wave * 16;
  const float* __restrict__ arow = A + (size_t)(rowbase + m16) * 128;

  v8f acc[8];
#pragma unroll
  for (int nt = 0; nt < 8; ++nt)
#pragma unroll
    for (int r = 0; r < 8; ++r) acc[nt][r] = 0.f;

#pragma unroll
  for (int kk = 0; kk < 4; ++kk) {                 // K = 128 in steps of 32
    const int kb = kk * 32 + hi * 8;
    v16bf a;
#pragma unroll
    for (int j = 0; j < 8; ++j) {                  // A 16x32 bf16 lane layout
      a[j]     = (__bf16)arow[kb + j];
      a[8 + j] = (__bf16)arow[kb + 16 + j];
    }
#pragma unroll
    for (int nt = 0; nt < 8; ++nt) {
      const __bf16* __restrict__ wp = &ldsw[(nt * 16 + m16) * 130 + kb];
      v16bf b;
#pragma unroll
      for (int j = 0; j < 8; ++j) {
        b[j]     = wp[j];
        b[8 + j] = wp[16 + j];
      }
      acc[nt] = __builtin_amdgcn_wmma_f32_16x16x32_bf16(
          false, a, false, b, (short)0, acc[nt], false, false);
    }
  }

#pragma unroll
  for (int nt = 0; nt < 8; ++nt) {
    const int col = nt * 16 + m16;
    const float bv = bias[col];
#pragma unroll
    for (int r = 0; r < 8; ++r) {                  // C/D layout: VGPR r -> row r + 8*hi
      const int orow = rowbase + r + hi * 8;
      out[(size_t)orow * 128 + col] = acc[nt][r] + bv;
    }
  }
}

// ---------- per-layer init ---------------------------------------------------
__global__ void k_init(float* __restrict__ agg, size_t aggn,
                       unsigned* __restrict__ nmax, float* __restrict__ nden) {
  const size_t idx = (size_t)blockIdx.x * blockDim.x + threadIdx.x;
  const size_t stride = (size_t)gridDim.x * blockDim.x;
  for (size_t i = idx; i < aggn; i += stride) agg[i] = 0.f;
  for (size_t i = idx; i < MM; i += stride) { nmax[i] = 0u; nden[i] = 0.f; }
}

// ---------- edge pass 1 (C=128): score + segment max, one wave per edge ------
__global__ void k_edge_score128(const int* __restrict__ ei,
                                const float* __restrict__ xl, const float* __restrict__ xr,
                                const float* __restrict__ att,
                                float* __restrict__ esc, unsigned* __restrict__ nmax) {
  const int gw   = (blockIdx.x * blockDim.x + threadIdx.x) >> 5;
  const int lane = threadIdx.x & 31;
  const int nw   = (gridDim.x * blockDim.x) >> 5;
  const float4 a4 = ((const float4*)att)[lane];
  for (int e = gw; e < ETOT; e += nw) {
    int s, t; edge_st(ei, e, s, t);
    const int en = e + nw;                         // prefetch next edge's rows
    if (en < ETOT) {
      int s2, t2; edge_st(ei, en, s2, t2);
      __builtin_prefetch(xl + (size_t)s2 * 128 + lane * 4, 0, 3);
      __builtin_prefetch(xr + (size_t)t2 * 128 + lane * 4, 0, 3);
    }
    const float4 l4 = ((const float4*)(xl + (size_t)s * 128))[lane];
    const float4 r4 = ((const float4*)(xr + (size_t)t * 128))[lane];
    float p = 0.f, v;
    v = l4.x + r4.x; p += a4.x * lrelu(v, 0.2f);
    v = l4.y + r4.y; p += a4.y * lrelu(v, 0.2f);
    v = l4.z + r4.z; p += a4.z * lrelu(v, 0.2f);
    v = l4.w + r4.w; p += a4.w * lrelu(v, 0.2f);
#pragma unroll
    for (int off = 16; off > 0; off >>= 1) p += __shfl_xor(p, off, 32);
    if (lane == 0) { esc[e] = p; atomicMax(&nmax[t], fkey(p)); }
  }
}

// ---------- edge pass 2: exp + segment sum (shared by C=128 and C=7) ---------
__global__ void k_edge_norm(const int* __restrict__ ei, const float* __restrict__ esc,
                            float* __restrict__ eex, const unsigned* __restrict__ nmax,
                            float* __restrict__ nden) {
  const int idx = blockIdx.x * blockDim.x + threadIdx.x;
  const int stride = gridDim.x * blockDim.x;
  for (int e = idx; e < ETOT; e += stride) {
    int s, t; edge_st(ei, e, s, t); (void)s;
    const float ex = __expf(esc[e] - funkey(nmax[t]));
    eex[e] = ex;
    atomicAdd(&nden[t], ex);
  }
}

// ---------- edge pass 3 (C=128): weighted scatter, one wave per edge ---------
__global__ void k_edge_aggr128(const int* __restrict__ ei, const float* __restrict__ xl,
                               const float* __restrict__ eex, const float* __restrict__ nden,
                               float* __restrict__ agg) {
  const int gw   = (blockIdx.x * blockDim.x + threadIdx.x) >> 5;
  const int lane = threadIdx.x & 31;
  const int nw   = (gridDim.x * blockDim.x) >> 5;
  for (int e = gw; e < ETOT; e += nw) {
    int s, t; edge_st(ei, e, s, t);
    const float alpha = eex[e] / (nden[t] + 1e-16f);
    const float4 l4 = ((const float4*)(xl + (size_t)s * 128))[lane];
    float* __restrict__ ap = agg + (size_t)t * 128 + lane * 4;
    atomicAdd(ap + 0, l4.x * alpha);
    atomicAdd(ap + 1, l4.y * alpha);
    atomicAdd(ap + 2, l4.z * alpha);
    atomicAdd(ap + 3, l4.w * alpha);
  }
}

// ---------- finalize: out = (agg + bias) with optional leaky relu 0.01 -------
__global__ void k_finalize128(const float* __restrict__ agg, const float* __restrict__ bias,
                              float* __restrict__ out, int act) {
  const size_t idx = (size_t)blockIdx.x * blockDim.x + threadIdx.x;
  const size_t stride = (size_t)gridDim.x * blockDim.x;
  for (size_t i = idx; i < (size_t)MM * 128; i += stride) {
    float v = agg[i] + bias[i & 127];
    out[i] = act ? lrelu(v, 0.01f) : v;
  }
}

// ---------- small GEMM for a2: out[M,8] (7 valid) = A[M,128] @ W[128,7] + b --
__global__ __launch_bounds__(128)
void k_gemm7(const float* __restrict__ A, const float* __restrict__ W,
             const float* __restrict__ b, float* __restrict__ out) {
  __shared__ float w[128 * CH];
  __shared__ float bb[CH];
  const int tid = threadIdx.x;
  for (int i = tid; i < 128 * CH; i += blockDim.x) w[i] = W[i];
  if (tid < CH) bb[tid] = b[tid];
  __syncthreads();
  const int idx = blockIdx.x * blockDim.x + tid;
  const int stride = gridDim.x * blockDim.x;
  for (int m = idx; m < MM; m += stride) {
    const float* __restrict__ ar = A + (size_t)m * 128;
    float acc[CH];
#pragma unroll
    for (int c = 0; c < CH; ++c) acc[c] = bb[c];
    for (int k = 0; k < 128; ++k) {
      const float av = ar[k];
#pragma unroll
      for (int c = 0; c < CH; ++c) acc[c] += av * w[k * CH + c];
    }
#pragma unroll
    for (int c = 0; c < CH; ++c) out[(size_t)m * 8 + c] = acc[c];
    out[(size_t)m * 8 + CH] = 0.f;
  }
}

// ---------- edge passes for C=7 (thread per edge) ----------------------------
__global__ void k_edge_score7(const int* __restrict__ ei,
                              const float* __restrict__ xl, const float* __restrict__ xr,
                              const float* __restrict__ att,
                              float* __restrict__ esc, unsigned* __restrict__ nmax) {
  const int idx = blockIdx.x * blockDim.x + threadIdx.x;
  const int stride = gridDim.x * blockDim.x;
  for (int e = idx; e < ETOT; e += stride) {
    int s, t; edge_st(ei, e, s, t);
    float p = 0.f;
#pragma unroll
    for (int c = 0; c < CH; ++c) {
      const float v = xl[(size_t)s * 8 + c] + xr[(size_t)t * 8 + c];
      p += att[c] * lrelu(v, 0.2f);
    }
    esc[e] = p;
    atomicMax(&nmax[t], fkey(p));
  }
}

__global__ void k_edge_aggr7(const int* __restrict__ ei, const float* __restrict__ xl,
                             const float* __restrict__ eex, const float* __restrict__ nden,
                             float* __restrict__ agg) {
  const int idx = blockIdx.x * blockDim.x + threadIdx.x;
  const int stride = gridDim.x * blockDim.x;
  for (int e = idx; e < ETOT; e += stride) {
    int s, t; edge_st(ei, e, s, t);
    const float alpha = eex[e] / (nden[t] + 1e-16f);
#pragma unroll
    for (int c = 0; c < CH; ++c)
      atomicAdd(&agg[(size_t)t * 8 + c], xl[(size_t)s * 8 + c] * alpha);
  }
}

// ---------- softmax over 7 logits per node -----------------------------------
__global__ void k_softmax7(const float* __restrict__ agg, const float* __restrict__ bias,
                           float* __restrict__ probs) {
  const int idx = blockIdx.x * blockDim.x + threadIdx.x;
  const int stride = gridDim.x * blockDim.x;
  for (int m = idx; m < MM; m += stride) {
    float lg[CH], mx = -1e30f;
#pragma unroll
    for (int c = 0; c < CH; ++c) { lg[c] = agg[(size_t)m * 8 + c] + bias[c]; mx = fmaxf(mx, lg[c]); }
    float sum = 0.f;
#pragma unroll
    for (int c = 0; c < CH; ++c) { lg[c] = __expf(lg[c] - mx); sum += lg[c]; }
    const float inv = 1.f / sum;
#pragma unroll
    for (int c = 0; c < CH; ++c) probs[(size_t)m * 8 + c] = lg[c] * inv;
    probs[(size_t)m * 8 + CH] = 0.f;
  }
}

// ---------- gumbel + top-4 of 7, 32 work items -------------------------------
__global__ void k_gumbel_topk(const float* __restrict__ probs, const float* __restrict__ gu,
                              float* __restrict__ out_action, float* __restrict__ out_sel) {
  const int t = threadIdx.x;
  if (t >= 32) return;
  const int b = t >> 3, col = t & 7;
  const int node = b * NNODE + col;
  float sc[CH];
  bool used[CH];
#pragma unroll
  for (int c = 0; c < CH; ++c) {
    const float p = probs[(size_t)node * 8 + c];
    const float g = -__logf(-__logf(gu[(b * 8 + col) * CH + c]));
    sc[c] = __logf(p) + g;
    used[c] = false;
  }
#pragma unroll
  for (int k = 0; k < 4; ++k) {                    // stable argmax, matches top_k tie rule
    int best = 0; float bv = -1e30f;
#pragma unroll
    for (int c = 0; c < CH; ++c)
      if (!used[c] && sc[c] > bv) { bv = sc[c]; best = c; }
    used[best] = true;
    out_action[(b * 8 + col) * 4 + k] = (float)best;
    out_sel[(b * 8 + col) * 4 + k]    = probs[((size_t)b * NNODE + best) * 8 + k];
  }
}

// ---------- value head: dot-128 per node -------------------------------------
__global__ __launch_bounds__(128)
void k_value(const float* __restrict__ h, const float* __restrict__ fcw,
             const float* __restrict__ fcb, float* __restrict__ out) {
  __shared__ float w[128];
  w[threadIdx.x] = fcw[threadIdx.x];
  __syncthreads();
  const float fb = fcb[0];
  const int idx = blockIdx.x * blockDim.x + threadIdx.x;
  const int stride = gridDim.x * blockDim.x;
  for (int m = idx; m < MM; m += stride) {
    const float* __restrict__ hr = h + (size_t)m * 128;
    float acc = fb;
#pragma unroll 4
    for (int k = 0; k < 128; ++k) acc += hr[k] * w[k];
    out[m] = acc;
  }
}

// ---------- host orchestration ----------------------------------------------
extern "C" void kernel_launch(void* const* d_in, const int* in_sizes, int n_in,
                              void* d_out, int out_size, void* d_ws, size_t ws_size,
                              hipStream_t stream) {
  (void)in_sizes; (void)n_in; (void)out_size; (void)ws_size;
  const float* x  = (const float*)d_in[0];
  const int*   ei = (const int*)d_in[1];
  const float* gu = (const float*)d_in[2];
  // per-layer params: Wl, bl, Wr, br, att, b at base..base+5
  const float** P = (const float**)d_in;
  const int A1 = 3, A2 = 9, C1 = 15, C2 = 21;
  const float* fcW = (const float*)d_in[27];
  const float* fcb = (const float*)d_in[28];

  float* ws = (float*)d_ws;
  size_t o = 0;
  float*    XL    = ws + o; o += (size_t)MM * 128;
  float*    XR    = ws + o; o += (size_t)MM * 128;
  float*    AGG   = ws + o; o += (size_t)MM * 128;
  float*    H     = ws + o; o += (size_t)MM * 128;
  float*    ESC   = ws + o; o += ETOT;
  float*    EEX   = ws + o; o += ETOT;
  unsigned* NMAX  = (unsigned*)(ws + o); o += MM;
  float*    NDEN  = ws + o; o += MM;
  float*    XL2   = ws + o; o += (size_t)MM * 8;
  float*    XR2   = ws + o; o += (size_t)MM * 8;
  float*    AGG2  = ws + o; o += (size_t)MM * 8;
  float*    PROBS = ws + o; o += (size_t)MM * 8;

  float* out_action = (float*)d_out;           // [4,8,4]
  float* out_sel    = out_action + 128;        // [4,8,4]
  float* out_value  = out_action + 256;        // [4,10000,1]

  const dim3 gThr(256);
  const dim3 gEdgeW(2048);                     // wave-per-edge kernels
  const dim3 gEdgeT(2048);                     // thread-per-edge kernels
  const dim3 gElem(2048);
  const dim3 gNode(512);

  auto big_layer = [&](const float* in, int pb, float* outbuf, int act) {
    k_gemm128<<<MM / 64, 128, 0, stream>>>(in, P[pb + 0], P[pb + 1], XL);
    k_gemm128<<<MM / 64, 128, 0, stream>>>(in, P[pb + 2], P[pb + 3], XR);
    k_init<<<gElem, gThr, 0, stream>>>(AGG, (size_t)MM * 128, NMAX, NDEN);
    k_edge_score128<<<gEdgeW, gThr, 0, stream>>>(ei, XL, XR, P[pb + 4], ESC, NMAX);
    k_edge_norm<<<gEdgeT, gThr, 0, stream>>>(ei, ESC, EEX, NMAX, NDEN);
    k_edge_aggr128<<<gEdgeW, gThr, 0, stream>>>(ei, XL, EEX, NDEN, AGG);
    k_finalize128<<<gElem, gThr, 0, stream>>>(AGG, P[pb + 5], outbuf, act);
  };

  // actor layer 1: x -> H (leaky 0.01)
  big_layer(x, A1, H, 1);

  // actor layer 2 (C=7): H -> logits -> probs
  k_gemm7<<<512, 128, 0, stream>>>(H, P[A2 + 0], P[A2 + 1], XL2);
  k_gemm7<<<512, 128, 0, stream>>>(H, P[A2 + 2], P[A2 + 3], XR2);
  k_init<<<gElem, gThr, 0, stream>>>(AGG2, (size_t)MM * 8, NMAX, NDEN);
  k_edge_score7<<<gEdgeT, gThr, 0, stream>>>(ei, XL2, XR2, P[A2 + 4], ESC, NMAX);
  k_edge_norm<<<gEdgeT, gThr, 0, stream>>>(ei, ESC, EEX, NMAX, NDEN);
  k_edge_aggr7<<<gEdgeT, gThr, 0, stream>>>(ei, XL2, EEX, NDEN, AGG2);
  k_softmax7<<<gNode, gThr, 0, stream>>>(AGG2, P[A2 + 5], PROBS);
  k_gumbel_topk<<<1, 32, 0, stream>>>(PROBS, gu, out_action, out_sel);

  // critic layer 1: x -> H (leaky 0.01); critic layer 2: H -> H (no act)
  big_layer(x, C1, H, 1);
  big_layer(H, C2, H, 0);

  // value head
  k_value<<<gNode, 128, 0, stream>>>(H, fcW, fcb, out_value);
}
